// Classifier_76227079570084
// MI455X (gfx1250) — compile-verified, matrix operand order checked by reading, not run on hardware
//
#include <hip/hip_runtime.h>

#define SD 127
#define NVOXC (127 * 127 * 127)

typedef __attribute__((ext_vector_type(16))) _Float16 v16h;
typedef __attribute__((ext_vector_type(8)))  _Float16 v8h;
typedef __attribute__((ext_vector_type(4)))  _Float16 v4h;
typedef __attribute__((ext_vector_type(8)))  float    v8f;

#if defined(__has_builtin)
#if __has_builtin(__builtin_amdgcn_ballot_w32)
#define BALLOT32(x) __builtin_amdgcn_ballot_w32(x)
#endif
#endif
#ifndef BALLOT32
#define BALLOT32(x) 1u
#endif

// Per-lane K indices of the f16 A/B fragment (16x16x32 WMMA):
// element e: K = base + (e<8 ? e : e+8), base = (lane&16) ? 8 : 0
// => halves [base, base+8) and [base+16, base+24): two 16B loads.
__device__ __forceinline__ v16h pack16(v8h lo, v8h hi) {
  v16h r;
#pragma unroll
  for (int e = 0; e < 8; ++e) { r[e] = lo[e]; r[e + 8] = hi[e]; }
  return r;
}
__device__ __forceinline__ v16h zero16() {
  v16h r;
#pragma unroll
  for (int e = 0; e < 16; ++e) r[e] = (_Float16)0.f;
  return r;
}

__global__ void k_zero(float* __restrict__ p, long long n) {
  long long i = (long long)blockIdx.x * blockDim.x + threadIdx.x;
  long long st = (long long)gridDim.x * blockDim.x;
  for (; i < n; i += st) p[i] = 0.f;
}

__global__ void k_scatter(const int* __restrict__ coords, const float* __restrict__ colors,
                          int n, float* __restrict__ feat, float* __restrict__ cnt) {
  int i = blockIdx.x * blockDim.x + threadIdx.x;
  if (i >= n) return;
  int x = coords[4 * i + 0], y = coords[4 * i + 1], z = coords[4 * i + 2], b = coords[4 * i + 3];
  int flat = ((b * SD + x) * SD + y) * SD + z;
  atomicAdd(&feat[3 * flat + 0], colors[3 * i + 0]);
  atomicAdd(&feat[3 * flat + 1], colors[3 * i + 1]);
  atomicAdd(&feat[3 * flat + 2], colors[3 * i + 2]);
  atomicAdd(&cnt[flat], 1.f);
}

// Average features at active voxels into padded f16 quads + active list + voxel->row map.
__global__ void k_compact(const float* __restrict__ feat, const float* __restrict__ cnt,
                          _Float16* __restrict__ x4, int* __restrict__ list,
                          int* __restrict__ map, int* __restrict__ counter) {
  int v = blockIdx.x * blockDim.x + threadIdx.x;
  if (v >= NVOXC) return;
  float c = cnt[v];
  if (c > 0.f) {
    float inv = 1.f / c;
    x4[4 * v + 0] = (_Float16)(feat[3 * v + 0] * inv);
    x4[4 * v + 1] = (_Float16)(feat[3 * v + 1] * inv);
    x4[4 * v + 2] = (_Float16)(feat[3 * v + 2] * inv);
    x4[4 * v + 3] = (_Float16)0.f;
    int j = atomicAdd(counter, 1);
    list[j] = v;
    map[v] = j + 1;       // 0 == inactive
  }
}

// Transpose conv weights [kd][kh][kw][Cin][Cout] f32 -> [tap][Cout][Cin] f16
__global__ void k_wprep(const float* __restrict__ W, _Float16* __restrict__ Wh,
                        int taps, int Cin, int Cout) {
  int i = blockIdx.x * blockDim.x + threadIdx.x;
  if (i >= taps * Cin * Cout) return;
  int ci = i % Cin; int r = i / Cin; int co = r % Cout; int t = r / Cout;
  Wh[i] = (_Float16)W[(size_t)(t * Cin + ci) * Cout + co];
}

// conv0 weights: [co][kk] f16, kk = tap*4 + ci (ci 0..2 real, 3 pad), padded to 128
__global__ void k_w0prep(const float* __restrict__ W0, _Float16* __restrict__ W0t) {
  int i = blockIdx.x * blockDim.x + threadIdx.x;
  if (i >= 32 * 128) return;
  int kk = i & 127; int co = i >> 7;
  int t = kk >> 2, ci = kk & 3;
  float v = (t < 27 && ci < 3) ? W0[(size_t)(t * 3 + ci) * 32 + co] : 0.f;
  W0t[i] = (_Float16)v;
}

// Submanifold conv0 (3->32) at active sites. K flattened tap-major x4, padded to 128.
__global__ void k_conv0_wmma(const _Float16* __restrict__ x4, const _Float16* __restrict__ W0t,
                             const int* __restrict__ list, const int* __restrict__ counter,
                             float* __restrict__ y0s, int ntM) {
  int wave = blockIdx.x * (blockDim.x >> 5) + (threadIdx.x >> 5);
  if (wave >= ntM * 2) return;                 // wave-uniform
  int lane = threadIdx.x & 31;
  int nact = counter[0];
  int mt = wave >> 1, nt = wave & 1;
  int gm = mt * 16 + (lane & 15);
  bool valid = gm < nact;
  int site = valid ? list[gm] : 0;
  int vz = site % SD; int r0 = site / SD; int vy = r0 % SD; int vx = r0 / SD;
  int tb = (lane & 16) ? 2 : 0;                // base/4
  const _Float16* wrow = W0t + (size_t)(nt * 16 + (lane & 15)) * 128 + ((lane & 16) ? 8 : 0);
  v8f acc = {0.f, 0.f, 0.f, 0.f, 0.f, 0.f, 0.f, 0.f};
#pragma unroll
  for (int c = 0; c < 4; ++c) {
    v16h a;
#pragma unroll
    for (int g = 0; g < 4; ++g) {              // 4 taps per lane, 4 channels each
      int tt = c * 8 + tb + (g < 2 ? g : g + 2);
      v4h q = {(_Float16)0.f, (_Float16)0.f, (_Float16)0.f, (_Float16)0.f};
      if (valid && tt < 27) {
        int kd = tt / 9, rm = tt - 9 * kd, kh = rm / 3, kw = rm - 3 * kh;
        int nd = vx + kd - 1, nh = vy + kh - 1, nw = vz + kw - 1;
        if ((unsigned)nd < SD && (unsigned)nh < SD && (unsigned)nw < SD)
          q = *(const v4h*)(x4 + (size_t)((nd * SD + nh) * SD + nw) * 4);
      }
      a[4 * g + 0] = q[0]; a[4 * g + 1] = q[1]; a[4 * g + 2] = q[2]; a[4 * g + 3] = q[3];
    }
    v16h b = pack16(*(const v8h*)(wrow + c * 32), *(const v8h*)(wrow + c * 32 + 16));
    acc = __builtin_amdgcn_wmma_f32_16x16x32_f16(false, a, false, b, (short)0, acc, false, false);
  }
  int n = nt * 16 + (lane & 15);
  int mb = mt * 16 + ((lane & 16) ? 8 : 0);
#pragma unroll
  for (int r = 0; r < 8; ++r) {
    int g = mb + r;
    if (g < nact) y0s[(size_t)g * 32 + n] = acc[r];
  }
}

// MaxPool 3^3 stride 2 over active sites only (127 -> 63), builds pooled mask.
__global__ void k_maxpool3(const int* __restrict__ map, const float* __restrict__ y0s,
                           float* __restrict__ y1, float* __restrict__ m1) {
  const int D = 63;
  int s = blockIdx.x * blockDim.x + threadIdx.x;
  if (s >= D * D * D) return;
  int w = s % D; int r = s / D; int h = r % D; int d = r / D;
  float best[32];
#pragma unroll
  for (int c = 0; c < 32; ++c) best[c] = -1e30f;
  bool any = false;
  for (int kd = 0; kd < 3; ++kd)
    for (int kh = 0; kh < 3; ++kh)
      for (int kw = 0; kw < 3; ++kw) {
        int id = 2 * d + kd, ih = 2 * h + kh, iw = 2 * w + kw;  // <= 126, in bounds
        int mp = map[(id * SD + ih) * SD + iw];
        if (mp > 0) {
          any = true;
          const float* row = y0s + (size_t)(mp - 1) * 32;
          for (int c = 0; c < 32; ++c) best[c] = fmaxf(best[c], row[c]);
        }
      }
  m1[s] = any ? 1.f : 0.f;
  for (int c = 0; c < 32; ++c) y1[(size_t)s * 32 + c] = any ? best[c] : 0.f;
}

// Masked BN statistics: stats[0..C-1]=sum(x*m), stats[C..2C-1]=sum(x^2*m), stats[2C]=sum(m)
__global__ void k_bn_stats(const float* __restrict__ x, const float* __restrict__ msk,
                           int sites, int C, float* __restrict__ stats) {
  __shared__ float sh[352];
  for (int i = threadIdx.x; i < 2 * C + 1; i += blockDim.x) sh[i] = 0.f;
  __syncthreads();
  int s = blockIdx.x * blockDim.x + threadIdx.x;
  if (s < sites) {
    float m = msk[s];
    if (m > 0.f) {
      atomicAdd(&sh[2 * C], m);
      const float* xs = x + (size_t)s * C;
      for (int c = 0; c < C; ++c) {
        float v = xs[c];
        atomicAdd(&sh[c], v * m);
        atomicAdd(&sh[C + c], v * v * m);
      }
    }
  }
  __syncthreads();
  for (int i = threadIdx.x; i < 2 * C + 1; i += blockDim.x) atomicAdd(&stats[i], sh[i]);
}

// BN-ReLU, masked; output written as f16 (consumed only by WMMA convs / final linear)
__global__ void k_bn_apply(const float* __restrict__ xin, _Float16* __restrict__ xout,
                           const float* __restrict__ msk, const float* __restrict__ g,
                           const float* __restrict__ b, const float* __restrict__ stats,
                           int sites, int C) {
  int s = blockIdx.x * blockDim.x + threadIdx.x;
  if (s >= sites) return;
  float m = msk[s];
  float cnt = stats[2 * C]; cnt = cnt > 1.f ? cnt : 1.f;
  for (int c = 0; c < C; ++c) {
    float mean = stats[c] / cnt;
    float var  = stats[C + c] / cnt - mean * mean;
    float y = (xin[(size_t)s * C + c] - mean) * rsqrtf(var + 1e-4f) * g[c] + b[c];
    y = y > 0.f ? y : 0.f;
    xout[(size_t)s * C + c] = (_Float16)(y * m);
  }
}

// Pooled mask: 2^3 window, stride 2
__global__ void k_pool_mask2(const float* __restrict__ mi, float* __restrict__ mo,
                             int Din, int Dout) {
  int s = blockIdx.x * blockDim.x + threadIdx.x;
  if (s >= Dout * Dout * Dout) return;
  int w = s % Dout; int r = s / Dout; int h = r % Dout; int d = r / Dout;
  float best = 0.f;
  for (int kd = 0; kd < 2; ++kd)
    for (int kh = 0; kh < 2; ++kh)
      for (int kw = 0; kw < 2; ++kw)
        best = fmaxf(best, mi[((2 * d + kd) * Din + 2 * h + kh) * Din + 2 * w + kw]);
  mo[s] = best;
}

// Masked dense 3D conv as WMMA GEMM, f16 in / f16 weights [tap][Cout][Cin] / f32 out.
// One wave: 16 sites x 32 out channels (two accumulators share the A fragment).
template <int KSZ, int STRIDE, int PAD, int CIN, int COUT>
__global__ void k_conv_wmma(const _Float16* __restrict__ in, const _Float16* __restrict__ Wh,
                            const float* __restrict__ omask, float* __restrict__ out,
                            int Din, int Dout) {
  constexpr int TAPS = KSZ * KSZ * KSZ;
  constexpr int NP = COUT / 32;
  int sites = Dout * Dout * Dout;
  int ntM = (sites + 15) >> 4;
  int wave = blockIdx.x * (blockDim.x >> 5) + (threadIdx.x >> 5);
  if (wave >= ntM * NP) return;                // wave-uniform
  int lane = threadIdx.x & 31;
  int mt = wave / NP, np = wave - mt * NP;
  int site = mt * 16 + (lane & 15);
  bool sv = site < sites;
  int d = 0, h = 0, w = 0;
  if (sv) { w = site % Dout; int r = site / Dout; h = r % Dout; d = r / Dout; }
  int hb = (lane & 16) ? 8 : 0;
  const _Float16* wbase = Wh + (size_t)(np * 32 + (lane & 15)) * CIN + hb;
  v8f acc0 = {0.f, 0.f, 0.f, 0.f, 0.f, 0.f, 0.f, 0.f};
  v8f acc1 = {0.f, 0.f, 0.f, 0.f, 0.f, 0.f, 0.f, 0.f};
#pragma unroll 1
  for (int t = 0; t < TAPS; ++t) {
    int kd = t / (KSZ * KSZ), rm = t - kd * KSZ * KSZ, kh = rm / KSZ, kw = rm - kh * KSZ;
    int nd = d * STRIDE + kd - PAD, nh = h * STRIDE + kh - PAD, nw = w * STRIDE + kw - PAD;
    bool ib = sv && (unsigned)nd < (unsigned)Din && (unsigned)nh < (unsigned)Din &&
              (unsigned)nw < (unsigned)Din;
    if (BALLOT32(ib) == 0u) continue;          // uniform tap skip (EXEC stays full)
    const _Float16* arow = in + (size_t)((nd * Din + nh) * Din + nw) * CIN + hb;
    const _Float16* wrow = wbase + (size_t)t * COUT * CIN;
#pragma unroll
    for (int c = 0; c < CIN; c += 32) {
      v16h a = ib ? pack16(*(const v8h*)(arow + c), *(const v8h*)(arow + c + 16)) : zero16();
      v16h b0 = pack16(*(const v8h*)(wrow + c), *(const v8h*)(wrow + c + 16));
      v16h b1 = pack16(*(const v8h*)(wrow + (size_t)16 * CIN + c),
                       *(const v8h*)(wrow + (size_t)16 * CIN + c + 16));
      acc0 = __builtin_amdgcn_wmma_f32_16x16x32_f16(false, a, false, b0, (short)0, acc0, false, false);
      acc1 = __builtin_amdgcn_wmma_f32_16x16x32_f16(false, a, false, b1, (short)0, acc1, false, false);
    }
  }
  int n = np * 32 + (lane & 15);
  int mb = mt * 16 + ((lane & 16) ? 8 : 0);
#pragma unroll
  for (int r = 0; r < 8; ++r) {
    int s2 = mb + r;
    if (s2 < sites) {
      float mm = omask[s2];
      out[(size_t)s2 * COUT + n]      = acc0[r] * mm;
      out[(size_t)s2 * COUT + n + 16] = acc1[r] * mm;
    }
  }
}

__global__ void k_add(const float* __restrict__ a, const float* __restrict__ b,
                      float* __restrict__ o, int n) {
  int i = blockIdx.x * blockDim.x + threadIdx.x;
  if (i < n) o[i] = a[i] + b[i];
}

__global__ void k_linear(const _Float16* __restrict__ xF, const float* __restrict__ Wp,
                         const float* __restrict__ bp, float* __restrict__ out) {
  int o = threadIdx.x;
  if (o >= 21) return;
  float acc = bp[o];
  for (int c = 0; c < 160; ++c) acc += (float)xF[c] * Wp[c * 21 + o];
  out[o] = acc;
}

extern "C" void kernel_launch(void* const* d_in, const int* in_sizes, int n_in,
                              void* d_out, int out_size, void* d_ws, size_t ws_size,
                              hipStream_t stream) {
  (void)n_in; (void)out_size; (void)ws_size;
  const int*   coords = (const int*)d_in[0];
  const float* colors = (const float*)d_in[1];
  const int N = in_sizes[1] / 3;
  int pi = 2;
  const float* W0 = (const float*)d_in[pi++];
  const float *g0[5], *b0[5], *Wsc[5], *Wa[5], *g1[5], *b1[5], *Wb[5],
              *g2[5], *b2[5], *Wc[5], *g3[5], *b3[5], *Wd[5];
  for (int k = 0; k < 5; ++k) {
    g0[k]  = (const float*)d_in[pi++]; b0[k] = (const float*)d_in[pi++];
    Wsc[k] = (const float*)d_in[pi++]; Wa[k] = (const float*)d_in[pi++];
    g1[k]  = (const float*)d_in[pi++]; b1[k] = (const float*)d_in[pi++];
    Wb[k]  = (const float*)d_in[pi++];
    g2[k]  = (const float*)d_in[pi++]; b2[k] = (const float*)d_in[pi++];
    Wc[k]  = (const float*)d_in[pi++];
    g3[k]  = (const float*)d_in[pi++]; b3[k] = (const float*)d_in[pi++];
    Wd[k]  = (const float*)d_in[pi++];
  }
  const float* gF = (const float*)d_in[pi++];
  const float* bF = (const float*)d_in[pi++];
  const float* Wp = (const float*)d_in[pi++];
  const float* bp = (const float*)d_in[pi++];

  static const int NIN[5]  = {32, 32, 64, 96, 128};
  static const int NOUT[5] = {32, 64, 96, 128, 160};

  // --- workspace carve (identical layout every call; deterministic) ---
  char* wptr = (char*)d_ws;
  auto take = [&](size_t bytes) -> void* {
    char* p = wptr; wptr += (bytes + 255) & ~(size_t)255; return (void*)p;
  };
  const int nvox = NVOXC;
  float*     feat = (float*)take((size_t)nvox * 3 * 4);
  float*     cnt  = (float*)take((size_t)nvox * 4);
  _Float16*  x4   = (_Float16*)take((size_t)nvox * 4 * 2);
  int*       map  = (int*)take((size_t)nvox * 4);
  int*       list = (int*)take((size_t)N * 4);
  int*       meta = (int*)take(256);
  _Float16*  W0t  = (_Float16*)take(32 * 128 * 2);
  _Float16 *WscH[5], *WaH[5], *WbH[5], *WcH[5], *WdH[5];
  for (int k = 0; k < 5; ++k) {
    int ni = NIN[k], no = NOUT[k];
    WscH[k] = (_Float16*)take((size_t)8 * ni * no * 2);
    WaH[k]  = (_Float16*)take((size_t)8 * ni * no * 2);
    WbH[k]  = (_Float16*)take((size_t)27 * no * no * 2);
    WcH[k]  = (_Float16*)take((size_t)27 * no * no * 2);
    WdH[k]  = (_Float16*)take((size_t)27 * no * no * 2);
  }
  float*    y0s = (float*)take((size_t)N * 32 * 4);
  const int D1S = 63 * 63 * 63;
  float*    XA  = (float*)take((size_t)D1S * 32 * 4);
  float*    MA  = (float*)take((size_t)D1S * 4);
  _Float16* TH  = (_Float16*)take((size_t)D1S * 32 * 2);
  const size_t SB = (size_t)31 * 31 * 31 * 32 * 4;
  float*    SCb = (float*)take(SB);
  float*    H1  = (float*)take(SB);
  float*    H2  = (float*)take(SB);
  float*    XB  = (float*)take(SB);
  float*    XC  = (float*)take(SB);
  _Float16* H1h = (_Float16*)take(SB / 2);
  _Float16* H2h = (_Float16*)take(SB / 2);
  float*    MB  = (float*)take((size_t)31 * 31 * 31 * 4);
  float*    MC  = (float*)take((size_t)31 * 31 * 31 * 4);
  float*    stats = (float*)take(512 * 4);

  dim3 blk(256);
  auto gz  = [](long long n) { long long b = (n + 255) / 256; return dim3((unsigned)(b > 4096 ? 4096 : b)); };
  auto g1d = [](long long n) { return dim3((unsigned)((n + 255) / 256)); };

  // 0) weight prep (f32 -> transposed f16)
  k_w0prep<<<g1d(32 * 128), blk, 0, stream>>>(W0, W0t);
  for (int k = 0; k < 5; ++k) {
    int ni = NIN[k], no = NOUT[k];
    k_wprep<<<g1d((long long)8 * ni * no), blk, 0, stream>>>(Wsc[k], WscH[k], 8, ni, no);
    k_wprep<<<g1d((long long)8 * ni * no), blk, 0, stream>>>(Wa[k],  WaH[k],  8, ni, no);
    k_wprep<<<g1d((long long)27 * no * no), blk, 0, stream>>>(Wb[k], WbH[k], 27, no, no);
    k_wprep<<<g1d((long long)27 * no * no), blk, 0, stream>>>(Wc[k], WcH[k], 27, no, no);
    k_wprep<<<g1d((long long)27 * no * no), blk, 0, stream>>>(Wd[k], WdH[k], 27, no, no);
  }

  // 1) voxelize + compact active set
  k_zero<<<gz((long long)nvox * 3), blk, 0, stream>>>(feat, (long long)nvox * 3);
  k_zero<<<gz(nvox), blk, 0, stream>>>(cnt, nvox);
  k_zero<<<gz(nvox), blk, 0, stream>>>((float*)map, nvox);
  k_zero<<<gz((long long)nvox * 2), blk, 0, stream>>>((float*)x4, (long long)nvox * 2);
  k_zero<<<1, 64, 0, stream>>>((float*)meta, 64);
  k_scatter<<<g1d(N), blk, 0, stream>>>(coords, colors, N, feat, cnt);
  k_compact<<<g1d(nvox), blk, 0, stream>>>(feat, cnt, x4, list, map, meta);

  // 2) submanifold conv0 (3->32) at active sites only (WMMA)
  int ntM0 = (N + 15) / 16;
  k_conv0_wmma<<<g1d((long long)ntM0 * 2 * 32), blk, 0, stream>>>(x4, W0t, list, meta, y0s, ntM0);

  // 3) maxpool 3^3 s2 over actives: 127 -> 63 dense
  k_maxpool3<<<g1d(D1S), blk, 0, stream>>>(map, y0s, XA, MA);

  // 4) SparseResNet stages
  float* x = XA; float* m = MA; int D = 63;
  for (int k = 0; k < 5; ++k) {
    int ni = NIN[k], no = NOUT[k];
    int E = (D - 1) / 2;
    int DS = D * D * D, ES = E * E * E;
    float* m2 = (m == MB) ? MC : MB;
    float* xn = (x == XB) ? XC : XB;
    int NP = no / 32, ntM = (ES + 15) / 16;
    dim3 gc = g1d((long long)ntM * NP * 32);

    auto conv2 = [&](const _Float16* inp, const _Float16* wh, float* outp) {
      switch (k) {
        case 0: k_conv_wmma<2,2,0,32,32 ><<<gc, blk, 0, stream>>>(inp, wh, m2, outp, D, E); break;
        case 1: k_conv_wmma<2,2,0,32,64 ><<<gc, blk, 0, stream>>>(inp, wh, m2, outp, D, E); break;
        case 2: k_conv_wmma<2,2,0,64,96 ><<<gc, blk, 0, stream>>>(inp, wh, m2, outp, D, E); break;
        case 3: k_conv_wmma<2,2,0,96,128><<<gc, blk, 0, stream>>>(inp, wh, m2, outp, D, E); break;
        default:k_conv_wmma<2,2,0,128,160><<<gc, blk, 0, stream>>>(inp, wh, m2, outp, D, E); break;
      }
    };
    auto conv3 = [&](const _Float16* inp, const _Float16* wh, float* outp) {
      switch (no) {
        case 32:  k_conv_wmma<3,1,1,32,32  ><<<gc, blk, 0, stream>>>(inp, wh, m2, outp, E, E); break;
        case 64:  k_conv_wmma<3,1,1,64,64  ><<<gc, blk, 0, stream>>>(inp, wh, m2, outp, E, E); break;
        case 96:  k_conv_wmma<3,1,1,96,96  ><<<gc, blk, 0, stream>>>(inp, wh, m2, outp, E, E); break;
        case 128: k_conv_wmma<3,1,1,128,128><<<gc, blk, 0, stream>>>(inp, wh, m2, outp, E, E); break;
        default:  k_conv_wmma<3,1,1,160,160><<<gc, blk, 0, stream>>>(inp, wh, m2, outp, E, E); break;
      }
    };

    // rep0: BN-ReLU(x) -> TH (f16)
    k_zero<<<1, 256, 0, stream>>>(stats, 2 * ni + 1);
    k_bn_stats<<<g1d(DS), blk, 0, stream>>>(x, m, DS, ni, stats);
    k_bn_apply<<<g1d(DS), blk, 0, stream>>>(x, TH, m, g0[k], b0[k], stats, DS, ni);
    k_pool_mask2<<<g1d(ES), blk, 0, stream>>>(m, m2, D, E);
    conv2(TH, WscH[k], SCb);
    conv2(TH, WaH[k],  H1);
    k_zero<<<1, 256, 0, stream>>>(stats, 2 * no + 1);
    k_bn_stats<<<g1d(ES), blk, 0, stream>>>(H1, m2, ES, no, stats);
    k_bn_apply<<<g1d(ES), blk, 0, stream>>>(H1, H1h, m2, g1[k], b1[k], stats, ES, no);
    conv3(H1h, WbH[k], H2);
    k_add<<<g1d((long long)ES * no), blk, 0, stream>>>(SCb, H2, xn, ES * no);
    // rep1: identity + BNReLU-conv3-BNReLU-conv3
    k_zero<<<1, 256, 0, stream>>>(stats, 2 * no + 1);
    k_bn_stats<<<g1d(ES), blk, 0, stream>>>(xn, m2, ES, no, stats);
    k_bn_apply<<<g1d(ES), blk, 0, stream>>>(xn, H1h, m2, g2[k], b2[k], stats, ES, no);
    conv3(H1h, WcH[k], H2);
    k_zero<<<1, 256, 0, stream>>>(stats, 2 * no + 1);
    k_bn_stats<<<g1d(ES), blk, 0, stream>>>(H2, m2, ES, no, stats);
    k_bn_apply<<<g1d(ES), blk, 0, stream>>>(H2, H2h, m2, g3[k], b3[k], stats, ES, no);
    conv3(H2h, WdH[k], H1);
    k_add<<<g1d((long long)ES * no), blk, 0, stream>>>(xn, H1, xn, ES * no);
    x = xn; m = m2; D = E;
  }

  // 5) final BN-ReLU at 1^3 (160 ch) + linear -> 21 logits
  k_zero<<<1, 256, 0, stream>>>(stats, 2 * 160 + 1);
  k_bn_stats<<<1, blk, 0, stream>>>(x, m, 1, 160, stats);
  k_bn_apply<<<1, blk, 0, stream>>>(x, TH, m, gF, bF, stats, 1, 160);
  k_linear<<<1, 32, 0, stream>>>(TH, Wp, bp, (float*)d_out);
}
